// GraphNetwork_78134045049078
// MI455X (gfx1250) — matrix operand order repaired and checked by reading, not compiled, not run on hardware
//
#include <hip/hip_runtime.h>
#include <hip/hip_bf16.h>

typedef _Float16 h16;
typedef __attribute__((ext_vector_type(16))) _Float16 v16h;
typedef __attribute__((ext_vector_type(8)))  float    v8f;

__device__ __forceinline__ float lrelu(float e) { return e > 0.0f ? e : 0.2f * e; }

// ---------------- utility kernels ----------------
__global__ void k_fill(float* p, int n, float v) {
    int i = blockIdx.x * blockDim.x + threadIdx.x;
    if (i < n) p[i] = v;
}

__global__ void k_cvt_h16(const float* __restrict__ s, h16* __restrict__ d, int n) {
    int i = blockIdx.x * blockDim.x + threadIdx.x;
    if (i < n) d[i] = (h16)s[i];
}

// W [K][M] f32 row-major -> Wt [M][K] f16
__global__ void k_transpose_h16(const float* __restrict__ W, h16* __restrict__ Wt, int K, int M) {
    int i = blockIdx.x * blockDim.x + threadIdx.x;
    if (i >= K * M) return;
    int k = i / M, m = i % M;
    Wt[(size_t)m * K + k] = (h16)W[i];
}

// ---------------- graph aggregation ----------------
__global__ void k_deg(const int* __restrict__ dst, int E, float* __restrict__ deg) {
    int i = blockIdx.x * blockDim.x + threadIdx.x;
    if (i < E) atomicAdd(&deg[dst[i]], 1.0f);
}

// wave-per-edge scatter-add: acc[dst] += X[src]  (X is f16, acc f32)
__global__ __launch_bounds__(256)
void k_scatter(const int* __restrict__ src, const int* __restrict__ dst, int E,
               const h16* __restrict__ X, int D, float* __restrict__ acc) {
    int e = blockIdx.x * 8 + (threadIdx.x >> 5);
    if (e >= E) return;
    int lane = threadIdx.x & 31;
    int s = src[e], d = dst[e];
    const h16* xs = X + (size_t)s * D;
    float* ap = acc + (size_t)d * D;
    for (int f = lane; f < D; f += 32)
        atomicAdd(&ap[f], (float)xs[f]);
}

__global__ void k_agg_fin(const float* __restrict__ acc, const float* __restrict__ deg,
                          int D, int total, h16* __restrict__ out) {
    int i = blockIdx.x * blockDim.x + threadIdx.x;
    if (i >= total) return;
    int n = i / D;
    out[i] = (h16)(acc[i] / fmaxf(deg[n], 1.0f));
}

// ---------------- WMMA GEMM:  C[N,M] = A1 @ B1 (+ A2 @ B2) ----------------
// A: f16 [N][K] row-major.  Bt: f16 [M][K] (pre-transposed weight).
// grid = (N/16, M/64), block = 128 (4 waves; wave w owns cols [wy*64+w*16, +16)).
__global__ __launch_bounds__(128)
void k_gemm(const h16* __restrict__ A1, const h16* __restrict__ B1t, int K1,
            const h16* __restrict__ A2, const h16* __restrict__ B2t, int K2,
            float* __restrict__ C, int M) {
    const int lane = threadIdx.x & 31;
    const int r0   = blockIdx.x * 16;
    const int m0   = blockIdx.y * 64 + (threadIdx.x >> 5) * 16;
    const int hi   = lane >> 4;
    const int row  = r0 + (lane & 15);
    const int col  = m0 + (lane & 15);
    v8f acc = {};
    for (int k0 = 0; k0 < K1; k0 += 32) {
        union { v16h v; uint4 u[2]; } a, b;
        const uint4* pa = (const uint4*)(A1 + (size_t)row * K1 + k0 + hi * 8);
        a.u[0] = pa[0]; a.u[1] = pa[2];               // K = hi*8 + {0..7} and +16
        const uint4* pb = (const uint4*)(B1t + (size_t)col * K1 + k0 + hi * 8);
        b.u[0] = pb[0]; b.u[1] = pb[2];               // K = hi*8 + {0..7} and +16
        acc = __builtin_amdgcn_wmma_f32_16x16x32_f16(false, a.v, false, b.v,
                                                     (short)0, acc, false, false);
    }
    if (A2) {
        for (int k0 = 0; k0 < K2; k0 += 32) {
            union { v16h v; uint4 u[2]; } a, b;
            const uint4* pa = (const uint4*)(A2 + (size_t)row * K2 + k0 + hi * 8);
            a.u[0] = pa[0]; a.u[1] = pa[2];
            const uint4* pb = (const uint4*)(B2t + (size_t)col * K2 + k0 + hi * 8);
            b.u[0] = pb[0]; b.u[1] = pb[2];
            acc = __builtin_amdgcn_wmma_f32_16x16x32_f16(false, a.v, false, b.v,
                                                         (short)0, acc, false, false);
        }
    }
    for (int r = 0; r < 8; r++)
        C[(size_t)(r0 + hi * 8 + r) * M + col] = acc[r];
}

// ---------------- batch norm ----------------
__global__ __launch_bounds__(256)
void k_bn_stats(const float* __restrict__ H, int N, int C,
                float* __restrict__ sum, float* __restrict__ sumsq) {
    int c = threadIdx.x % C, sub = threadIdx.x / C, step = 256 / C;
    int r0 = blockIdx.x * 120;
    int rend = min(r0 + 120, N);
    float s = 0.0f, q = 0.0f;
    for (int r = r0 + sub; r < rend; r += step) {
        float v = H[(size_t)r * C + c];
        s += v; q += v * v;
    }
    atomicAdd(&sum[c], s);
    atomicAdd(&sumsq[c], q);
}

__global__ void k_bn_fin(const float* sum, const float* sumsq, const float* g,
                         const float* be, int N, int C, float* scale, float* shift) {
    int c = blockIdx.x * blockDim.x + threadIdx.x;
    if (c >= C) return;
    float mu  = sum[c] / (float)N;
    float var = sumsq[c] / (float)N - mu * mu;
    float inv = rsqrtf(var + 1e-5f);
    float sc  = g[c] * inv;
    scale[c] = sc;
    shift[c] = be[c] - mu * sc;
}

__global__ void k_bn_apply(const float* __restrict__ H, const float* __restrict__ scale,
                           const float* __restrict__ shift, int C, int total,
                           h16* __restrict__ Y) {
    int i = blockIdx.x * blockDim.x + threadIdx.x;
    if (i >= total) return;
    int c = i % C;
    float y = H[i] * scale[c] + shift[c];
    Y[i] = (h16)(y > 0.0f ? y : 0.0f);
}

// ---------------- GAT attention ----------------
// per (node, head): a_src/a_dst dot products over 64 channels
__global__ void k_att_scores(const float* __restrict__ xp, const float* __restrict__ ws,
                             const float* __restrict__ wd, float* __restrict__ as,
                             float* __restrict__ ad, int N) {
    int idx = blockIdx.x * blockDim.x + threadIdx.x;
    if (idx >= N * 4) return;
    int n = idx >> 2, h = idx & 3;
    const float* v = xp + (size_t)n * 256 + h * 64;
    const float* s = ws + h * 64;
    const float* d = wd + h * 64;
    float a = 0.0f, b = 0.0f;
#pragma unroll
    for (int k = 0; k < 64; k++) { a += v[k] * s[k]; b += v[k] * d[k]; }
    as[idx] = a; ad[idx] = b;
}

__global__ void k_att_max(const int* __restrict__ src, const int* __restrict__ dst,
                          int E, int N, const float* __restrict__ as,
                          const float* __restrict__ ad, float* __restrict__ m) {
    int idx = blockIdx.x * blockDim.x + threadIdx.x;
    if (idx >= E + N) return;
    int s, d;
    if (idx < E) { s = src[idx]; d = dst[idx]; } else { s = idx - E; d = s; }
#pragma unroll
    for (int h = 0; h < 4; h++) {
        float e = lrelu(as[s * 4 + h] + ad[d * 4 + h]);
        atomicMax(&m[d * 4 + h], e);
    }
}

__global__ void k_att_den(const int* __restrict__ src, const int* __restrict__ dst,
                          int E, int N, const float* __restrict__ as,
                          const float* __restrict__ ad, const float* __restrict__ m,
                          float* __restrict__ den) {
    int idx = blockIdx.x * blockDim.x + threadIdx.x;
    if (idx >= E + N) return;
    int s, d;
    if (idx < E) { s = src[idx]; d = dst[idx]; } else { s = idx - E; d = s; }
#pragma unroll
    for (int h = 0; h < 4; h++) {
        float e = lrelu(as[s * 4 + h] + ad[d * 4 + h]);
        atomicAdd(&den[d * 4 + h], expf(e - m[d * 4 + h]));
    }
}

// wave per edge: out[d] += (1/H) * sum_h alpha_h * xp[s,h,:]
__global__ __launch_bounds__(256)
void k_att_accum(const int* __restrict__ src, const int* __restrict__ dst, int E, int N,
                 const float* __restrict__ as, const float* __restrict__ ad,
                 const float* __restrict__ m, const float* __restrict__ den,
                 const float* __restrict__ xp, float* __restrict__ out) {
    int idx = blockIdx.x * 8 + (threadIdx.x >> 5);
    if (idx >= E + N) return;
    int lane = threadIdx.x & 31;
    int s, d;
    if (idx < E) { s = src[idx]; d = dst[idx]; } else { s = idx - E; d = s; }
    float al[4];
#pragma unroll
    for (int h = 0; h < 4; h++) {
        float e = lrelu(as[s * 4 + h] + ad[d * 4 + h]);
        al[h] = expf(e - m[d * 4 + h]) / den[d * 4 + h];
    }
    const float* xs = xp + (size_t)s * 256;
    float* op = out + (size_t)d * 64;
    for (int f = lane; f < 64; f += 32) {
        float v = al[0] * xs[f] + al[1] * xs[64 + f] + al[2] * xs[128 + f] + al[3] * xs[192 + f];
        atomicAdd(&op[f], 0.25f * v);
    }
}

// final: d_out[c] = mean_n relu(acc[n,c] + bg[c])
__global__ __launch_bounds__(256)
void k_final(const float* __restrict__ acc, const float* __restrict__ bg, int N,
             float* __restrict__ out) {
    int c = threadIdx.x & 63, sub = threadIdx.x >> 6;
    int r0 = blockIdx.x * 240;
    int rend = min(r0 + 240, N);
    float s = 0.0f;
    for (int r = r0 + sub; r < rend; r += 4) {
        float v = acc[(size_t)r * 64 + c] + bg[c];
        s += v > 0.0f ? v : 0.0f;
    }
    atomicAdd(&out[c], s * (1.0f / (float)N));
}

#define G1(n) dim3((unsigned)(((n) + 255) / 256)), dim3(256)

extern "C" void kernel_launch(void* const* d_in, const int* in_sizes, int n_in,
                              void* d_out, int out_size, void* d_ws, size_t ws_size,
                              hipStream_t stream) {
    (void)n_in; (void)out_size; (void)ws_size;
    const float* x    = (const float*)d_in[0];
    const int*   ei   = (const int*)d_in[1];
    const float* W1l  = (const float*)d_in[2];
    const float* W1r  = (const float*)d_in[4];
    const float* W2l  = (const float*)d_in[5];
    const float* W2r  = (const float*)d_in[7];
    const float* g1   = (const float*)d_in[8];
    const float* be1  = (const float*)d_in[9];
    const float* g2   = (const float*)d_in[10];
    const float* be2  = (const float*)d_in[11];
    const float* Wg   = (const float*)d_in[12];
    const float* atts = (const float*)d_in[13];
    const float* attd = (const float*)d_in[14];
    const float* bg   = (const float*)d_in[15];

    const int N = in_sizes[0] / 128;   // 30000 (divisible by 16)
    const int E = in_sizes[1] / 2;     // 300000
    const int* src = ei;
    const int* dst = ei + E;

    char* ws = (char*)d_ws;
    size_t off = 0;
    auto alloc = [&](size_t bytes) -> void* {
        void* p = ws + off;
        off += (bytes + 255) & ~(size_t)255;
        return p;
    };
    h16*   xh   = (h16*)  alloc((size_t)N * 128 * 2);   // x in f16
    float* aggf = (float*)alloc((size_t)N * 256 * 4);   // agg accum (reused L1/L2)
    h16*   aggh = (h16*)  alloc((size_t)N * 256 * 2);   // agg f16 (reused)
    float* hbuf = (float*)alloc((size_t)N * 256 * 4);   // h1, later xp
    h16*   h1rh = (h16*)  alloc((size_t)N * 256 * 2);
    float* h2   = (float*)alloc((size_t)N * 64 * 4);
    h16*   h2rh = (h16*)  alloc((size_t)N * 64 * 2);
    float* outa = (float*)alloc((size_t)N * 64 * 4);
    float* deg  = (float*)alloc((size_t)N * 4);
    float* asn  = (float*)alloc((size_t)N * 16);
    float* adn  = (float*)alloc((size_t)N * 16);
    float* mm   = (float*)alloc((size_t)N * 16);
    float* den  = (float*)alloc((size_t)N * 16);
    float* bsum = (float*)alloc(256 * 4);
    float* bsq  = (float*)alloc(256 * 4);
    float* bsc  = (float*)alloc(256 * 4);
    float* bsh  = (float*)alloc(256 * 4);
    h16* w1lt = (h16*)alloc(128 * 256 * 2);
    h16* w1rt = (h16*)alloc(128 * 256 * 2);
    h16* w2lt = (h16*)alloc(256 * 64 * 2);
    h16* w2rt = (h16*)alloc(256 * 64 * 2);
    h16* wgt  = (h16*)alloc(64 * 256 * 2);

    // ---- prep ----
    k_fill<<<G1(N), 0, stream>>>(deg, N, 0.0f);
    k_fill<<<G1(N * 128), 0, stream>>>(aggf, N * 128, 0.0f);
    k_cvt_h16<<<G1(N * 128), 0, stream>>>(x, xh, N * 128);
    k_transpose_h16<<<G1(128 * 256), 0, stream>>>(W1l, w1lt, 128, 256);
    k_transpose_h16<<<G1(128 * 256), 0, stream>>>(W1r, w1rt, 128, 256);
    k_transpose_h16<<<G1(256 * 64), 0, stream>>>(W2l, w2lt, 256, 64);
    k_transpose_h16<<<G1(256 * 64), 0, stream>>>(W2r, w2rt, 256, 64);
    k_transpose_h16<<<G1(64 * 256), 0, stream>>>(Wg, wgt, 64, 256);
    k_deg<<<G1(E), 0, stream>>>(dst, E, deg);

    // ---- SAGE layer 1: h1 = mean_agg(x)@W1l + x@W1r  (bias cancels in BN) ----
    k_scatter<<<dim3((E + 7) / 8), dim3(256), 0, stream>>>(src, dst, E, xh, 128, aggf);
    k_agg_fin<<<G1(N * 128), 0, stream>>>(aggf, deg, 128, N * 128, aggh);
    k_gemm<<<dim3(N / 16, 4), dim3(128), 0, stream>>>(aggh, w1lt, 128, xh, w1rt, 128, hbuf, 256);
    // BN1 + ReLU -> f16
    k_fill<<<dim3(1), dim3(256), 0, stream>>>(bsum, 256, 0.0f);
    k_fill<<<dim3(1), dim3(256), 0, stream>>>(bsq, 256, 0.0f);
    k_bn_stats<<<dim3((N + 119) / 120), dim3(256), 0, stream>>>(hbuf, N, 256, bsum, bsq);
    k_bn_fin<<<dim3(1), dim3(256), 0, stream>>>(bsum, bsq, g1, be1, N, 256, bsc, bsh);
    k_bn_apply<<<G1(N * 256), 0, stream>>>(hbuf, bsc, bsh, 256, N * 256, h1rh);

    // ---- SAGE layer 2 ----
    k_fill<<<G1(N * 256), 0, stream>>>(aggf, N * 256, 0.0f);
    k_scatter<<<dim3((E + 7) / 8), dim3(256), 0, stream>>>(src, dst, E, h1rh, 256, aggf);
    k_agg_fin<<<G1(N * 256), 0, stream>>>(aggf, deg, 256, N * 256, aggh);
    k_gemm<<<dim3(N / 16, 1), dim3(128), 0, stream>>>(aggh, w2lt, 256, h1rh, w2rt, 256, h2, 64);
    // BN2 + ReLU -> f16
    k_fill<<<dim3(1), dim3(64), 0, stream>>>(bsum, 64, 0.0f);
    k_fill<<<dim3(1), dim3(64), 0, stream>>>(bsq, 64, 0.0f);
    k_bn_stats<<<dim3((N + 119) / 120), dim3(256), 0, stream>>>(h2, N, 64, bsum, bsq);
    k_bn_fin<<<dim3(1), dim3(64), 0, stream>>>(bsum, bsq, g2, be2, N, 64, bsc, bsh);
    k_bn_apply<<<G1(N * 64), 0, stream>>>(h2, bsc, bsh, 64, N * 64, h2rh);

    // ---- GAT: xp = h2r @ Wg  ([N,256] = [N,4 heads,64]) ----
    k_gemm<<<dim3(N / 16, 4), dim3(128), 0, stream>>>(h2rh, wgt, 64, (const h16*)nullptr,
                                                      (const h16*)nullptr, 0, hbuf, 256);
    k_att_scores<<<G1(N * 4), 0, stream>>>(hbuf, atts, attd, asn, adn, N);
    k_fill<<<G1(N * 4), 0, stream>>>(mm, N * 4, -3.0e38f);
    k_fill<<<G1(N * 4), 0, stream>>>(den, N * 4, 0.0f);
    k_att_max<<<G1(E + N), 0, stream>>>(src, dst, E, N, asn, adn, mm);
    k_att_den<<<G1(E + N), 0, stream>>>(src, dst, E, N, asn, adn, mm, den);
    k_fill<<<G1(N * 64), 0, stream>>>(outa, N * 64, 0.0f);
    k_att_accum<<<dim3((E + N + 7) / 8), dim3(256), 0, stream>>>(src, dst, E, N, asn, adn,
                                                                 mm, den, hbuf, outa);

    // ---- final mean over nodes ----
    k_fill<<<dim3(1), dim3(64), 0, stream>>>((float*)d_out, 64, 0.0f);
    k_final<<<dim3((N + 239) / 240), dim3(256), 0, stream>>>(outa, bg, N, (float*)d_out);
}
